// CausalSelfAttention_15272903705390
// MI455X (gfx1250) — compile-verified, matrix operand order checked by reading, not run on hardware
//
#include <hip/hip_runtime.h>
#include <hip/hip_bf16.h>

// ---------------------------------------------------------------------------
// Problem constants (from reference): B=2, T=2048, C=1024, H=16, HD=64
// ---------------------------------------------------------------------------
#define BB 2
#define TT 2048
#define CC 1024
#define HH 16
#define HD 64
#define BT (BB * TT)          // 4096 token rows
#define C3 (3 * CC)           // 3072

typedef __attribute__((ext_vector_type(16))) __bf16 bf16x16;
typedef __attribute__((ext_vector_type(8)))  float  floatx8;

struct __attribute__((aligned(16))) bfv8 { __bf16 h[8]; };

// Build a 16-bit A/B WMMA operand for this lane.
// p must already point at:  rowBase + kbase + (lane>=16 ? 8 : 0)
// Layout (ISA 7.12.2, 16-bit A 16x32): lane<16 -> elems {K+0..7, K+16..23},
// lane>=16 -> elems {K+8..15, K+24..31}. Two 16B loads per lane.
__device__ inline bf16x16 load_op(const __bf16* p) {
    bfv8 lo = *reinterpret_cast<const bfv8*>(p);
    bfv8 hi = *reinterpret_cast<const bfv8*>(p + 16);
    bf16x16 v;
#pragma unroll
    for (int i = 0; i < 8; ++i) { v[i] = lo.h[i]; v[i + 8] = hi.h[i]; }
    return v;
}

__device__ inline floatx8 wmma_bf16(bf16x16 a, bf16x16 b, floatx8 c) {
    return __builtin_amdgcn_wmma_f32_16x16x32_bf16(
        /*neg_a=*/false, a, /*neg_b=*/false, b,
        /*c_mod=*/(short)0, c, /*reuse_a=*/false, /*reuse_b=*/false);
}

// ---------------------------------------------------------------------------
// fp32 -> bf16 conversion
// ---------------------------------------------------------------------------
__global__ void cvt_bf16_kernel(const float* __restrict__ src,
                                __bf16* __restrict__ dst, int n) {
    int i = blockIdx.x * blockDim.x + threadIdx.x;
    if (i < n) dst[i] = (__bf16)src[i];
}

// ---------------------------------------------------------------------------
// QKV GEMM: qkv[BT,3C] = xb[BT,C] @ Wab[3C,C]^T + b_attn
// One wave computes a 16(M) x 64(N) tile (4 accumulators, A reused 4x).
// Results scattered to q[B,H,T,HD], k[B,H,T,HD], vT[B,H,HD,T] in bf16.
// ---------------------------------------------------------------------------
__global__ void qkv_gemm_kernel(const __bf16* __restrict__ X,
                                const __bf16* __restrict__ W,
                                const float* __restrict__ bias,
                                __bf16* __restrict__ Q,
                                __bf16* __restrict__ K,
                                __bf16* __restrict__ VT) {
    const int lane = threadIdx.x & 31;
    const int wave = blockIdx.x * (blockDim.x >> 5) + (threadIdx.x >> 5);
    const int mtiles = BT / 16;           // 256
    const int nsup   = C3 / 64;           // 48
    const int mi = wave % mtiles;
    const int ni = wave / mtiles;
    if (ni >= nsup) return;

    const int m0 = mi * 16;
    const int n0 = ni * 64;
    const int ln = lane & 15;
    const int hi = lane >> 4;             // 0 or 1

    const long arow = (long)(m0 + ln) * CC + hi * 8;
    floatx8 acc0 = {}, acc1 = {}, acc2 = {}, acc3 = {};

    for (int k0 = 0; k0 < CC; k0 += 32) {
        bf16x16 a = load_op(X + arow + k0);
        bf16x16 b0 = load_op(W + (long)(n0 +  0 + ln) * CC + k0 + hi * 8);
        bf16x16 b1 = load_op(W + (long)(n0 + 16 + ln) * CC + k0 + hi * 8);
        bf16x16 b2 = load_op(W + (long)(n0 + 32 + ln) * CC + k0 + hi * 8);
        bf16x16 b3 = load_op(W + (long)(n0 + 48 + ln) * CC + k0 + hi * 8);
        acc0 = wmma_bf16(a, b0, acc0);
        acc1 = wmma_bf16(a, b1, acc1);
        acc2 = wmma_bf16(a, b2, acc2);
        acc3 = wmma_bf16(a, b3, acc3);
    }

    floatx8 accs[4] = {acc0, acc1, acc2, acc3};
#pragma unroll
    for (int t = 0; t < 4; ++t) {
        const int n = n0 + t * 16 + ln;   // output dim in [0,3C)
        const float bia = bias[n];
        const int region = n / CC;        // 0=q 1=k 2=v
        const int d = n % CC;
        const int h = d >> 6;
        const int hd = d & 63;
#pragma unroll
        for (int r = 0; r < 8; ++r) {
            const int mrow = hi ? (r + 8) : r;     // C-layout row mapping
            const int trow = m0 + mrow;            // token row in [0,BT)
            const int bidx = trow / TT;
            const int tok  = trow % TT;
            const __bf16 hv = (__bf16)(accs[t][r] + bia);
            const long bhbase = ((long)bidx * HH + h);
            if (region == 0)      Q[(bhbase * TT + tok) * HD + hd] = hv;
            else if (region == 1) K[(bhbase * TT + tok) * HD + hd] = hv;
            else                  VT[(bhbase * HD + hd) * TT + tok] = hv;
        }
    }
}

// ---------------------------------------------------------------------------
// Flash attention: one wave per (b,h, 16-query tile). Online softmax over
// 32-key chunks; P transposed C-layout -> A-layout through per-wave LDS.
// ---------------------------------------------------------------------------
__global__ void attn_kernel(const __bf16* __restrict__ Q,
                            const __bf16* __restrict__ K,
                            const __bf16* __restrict__ VT,
                            __bf16* __restrict__ Y) {
    __shared__ float pbuf[4][16 * 32];    // per-wave P tile (fp32)

    const int lane = threadIdx.x & 31;
    const int wid  = threadIdx.x >> 5;
    const int wave = blockIdx.x * 4 + wid;
    const int qtiles = TT / 16;           // 128
    const int qt = wave % qtiles;
    const int bh = wave / qtiles;         // b*H + h, in [0, 32)
    if (bh >= BB * HH) return;

    const __bf16* q  = Q  + (long)bh * TT * HD;
    const __bf16* k  = K  + (long)bh * TT * HD;
    const __bf16* vt = VT + (long)bh * HD * TT;

    const int ln = lane & 15;
    const int hi = lane >> 4;
    const int qrow = qt * 16 + ln;

    // Q A-operands (16 x 64) live in registers the whole kernel.
    const bf16x16 qa0 = load_op(q + (long)qrow * HD +  0 + hi * 8);
    const bf16x16 qa1 = load_op(q + (long)qrow * HD + 32 + hi * 8);

    floatx8 yacc0 = {}, yacc1 = {}, yacc2 = {}, yacc3 = {};
    float m[8], l[8];
#pragma unroll
    for (int r = 0; r < 8; ++r) { m[r] = -1e30f; l[r] = 0.0f; }

    const float scale = 0.125f;           // 1/sqrt(64)
    float* P = pbuf[wid];

    for (int t0 = 0; t0 < TT; t0 += 32) {
        // ---- S = (Q . K^T) * scale, two 16x16 tiles (keys t0..15, t0+16..31)
        floatx8 s0 = {}, s1 = {};
        {
            bf16x16 kb0 = load_op(k + (long)(t0 + ln) * HD +  0 + hi * 8);
            bf16x16 kb1 = load_op(k + (long)(t0 + ln) * HD + 32 + hi * 8);
            s0 = wmma_bf16(qa0, kb0, s0);
            s0 = wmma_bf16(qa1, kb1, s0);
            bf16x16 kc0 = load_op(k + (long)(t0 + 16 + ln) * HD +  0 + hi * 8);
            bf16x16 kc1 = load_op(k + (long)(t0 + 16 + ln) * HD + 32 + hi * 8);
            s1 = wmma_bf16(qa0, kc0, s1);
            s1 = wmma_bf16(qa1, kc1, s1);
        }
#pragma unroll
        for (int r = 0; r < 8; ++r) { s0[r] *= scale; s1[r] *= scale; }

        // ---- row max (butterfly within each 16-lane half == row group)
        float mx[8];
#pragma unroll
        for (int r = 0; r < 8; ++r) mx[r] = fmaxf(s0[r], s1[r]);
#pragma unroll
        for (int off = 1; off < 16; off <<= 1) {
#pragma unroll
            for (int r = 0; r < 8; ++r)
                mx[r] = fmaxf(mx[r], __shfl_xor(mx[r], off, 32));
        }

        float alpha[8];
#pragma unroll
        for (int r = 0; r < 8; ++r) {
            const float mnew = fmaxf(m[r], mx[r]);
            alpha[r] = __expf(m[r] - mnew);
            m[r] = mnew;
            s0[r] = __expf(s0[r] - mnew);
            s1[r] = __expf(s1[r] - mnew);
        }

        // ---- row sum
        float sm[8];
#pragma unroll
        for (int r = 0; r < 8; ++r) sm[r] = s0[r] + s1[r];
#pragma unroll
        for (int off = 1; off < 16; off <<= 1) {
#pragma unroll
            for (int r = 0; r < 8; ++r)
                sm[r] += __shfl_xor(sm[r], off, 32);
        }
#pragma unroll
        for (int r = 0; r < 8; ++r) l[r] = l[r] * alpha[r] + sm[r];

        // ---- rescale running Y
#pragma unroll
        for (int r = 0; r < 8; ++r) {
            yacc0[r] *= alpha[r]; yacc1[r] *= alpha[r];
            yacc2[r] *= alpha[r]; yacc3[r] *= alpha[r];
        }

        // ---- transpose P: C-layout regs -> LDS row-major [16 q][32 keys]
#pragma unroll
        for (int r = 0; r < 8; ++r) {
            const int row = hi ? (r + 8) : r;
            P[row * 32 + ln]      = s0[r];
            P[row * 32 + 16 + ln] = s1[r];
        }
        __syncthreads();

        // ---- read back as bf16 A operand (16x32)
        bf16x16 pa;
#pragma unroll
        for (int i = 0; i < 8; ++i) {
            pa[i]     = (__bf16)P[ln * 32 + hi * 8 + i];
            pa[i + 8] = (__bf16)P[ln * 32 + 16 + hi * 8 + i];
        }
        __syncthreads();

        // ---- Y += P . V   (B operand from vT: rows = head dim, K = keys)
        bf16x16 vb0 = load_op(vt + (long)( 0 + ln) * TT + t0 + hi * 8);
        bf16x16 vb1 = load_op(vt + (long)(16 + ln) * TT + t0 + hi * 8);
        bf16x16 vb2 = load_op(vt + (long)(32 + ln) * TT + t0 + hi * 8);
        bf16x16 vb3 = load_op(vt + (long)(48 + ln) * TT + t0 + hi * 8);
        yacc0 = wmma_bf16(pa, vb0, yacc0);
        yacc1 = wmma_bf16(pa, vb1, yacc1);
        yacc2 = wmma_bf16(pa, vb2, yacc2);
        yacc3 = wmma_bf16(pa, vb3, yacc3);
    }

    // ---- finalize: Y /= l, store bf16 into y[B,T,C]
    const int b_ = bh / HH;
    const int h_ = bh % HH;
    floatx8 ys[4] = {yacc0, yacc1, yacc2, yacc3};
#pragma unroll
    for (int nt = 0; nt < 4; ++nt) {
#pragma unroll
        for (int r = 0; r < 8; ++r) {
            const int row = hi ? (r + 8) : r;
            const int tglob = qt * 16 + row;
            const float val = ys[nt][r] / l[r];
            Y[((long)b_ * TT + tglob) * CC + h_ * HD + nt * 16 + ln] = (__bf16)val;
        }
    }
}

// ---------------------------------------------------------------------------
// Output projection: out[BT,C] = y[BT,C] @ Wproj[C,C]^T + b_proj  (fp32 out)
// ---------------------------------------------------------------------------
__global__ void proj_gemm_kernel(const __bf16* __restrict__ Yb,
                                 const __bf16* __restrict__ W,
                                 const float* __restrict__ bias,
                                 float* __restrict__ out) {
    const int lane = threadIdx.x & 31;
    const int wave = blockIdx.x * (blockDim.x >> 5) + (threadIdx.x >> 5);
    const int mtiles = BT / 16;           // 256
    const int nsup   = CC / 64;           // 16
    const int mi = wave % mtiles;
    const int ni = wave / mtiles;
    if (ni >= nsup) return;

    const int m0 = mi * 16;
    const int n0 = ni * 64;
    const int ln = lane & 15;
    const int hi = lane >> 4;

    const long arow = (long)(m0 + ln) * CC + hi * 8;
    floatx8 acc0 = {}, acc1 = {}, acc2 = {}, acc3 = {};

    for (int k0 = 0; k0 < CC; k0 += 32) {
        bf16x16 a = load_op(Yb + arow + k0);
        bf16x16 b0 = load_op(W + (long)(n0 +  0 + ln) * CC + k0 + hi * 8);
        bf16x16 b1 = load_op(W + (long)(n0 + 16 + ln) * CC + k0 + hi * 8);
        bf16x16 b2 = load_op(W + (long)(n0 + 32 + ln) * CC + k0 + hi * 8);
        bf16x16 b3 = load_op(W + (long)(n0 + 48 + ln) * CC + k0 + hi * 8);
        acc0 = wmma_bf16(a, b0, acc0);
        acc1 = wmma_bf16(a, b1, acc1);
        acc2 = wmma_bf16(a, b2, acc2);
        acc3 = wmma_bf16(a, b3, acc3);
    }

    floatx8 accs[4] = {acc0, acc1, acc2, acc3};
#pragma unroll
    for (int t = 0; t < 4; ++t) {
        const int n = n0 + t * 16 + ln;
        const float bia = bias[n];
#pragma unroll
        for (int r = 0; r < 8; ++r) {
            const int mrow = hi ? (r + 8) : r;
            out[(long)(m0 + mrow) * CC + n] = accs[t][r] + bia;
        }
    }
}

// ---------------------------------------------------------------------------
// Host launcher
// ---------------------------------------------------------------------------
extern "C" void kernel_launch(void* const* d_in, const int* in_sizes, int n_in,
                              void* d_out, int out_size, void* d_ws, size_t ws_size,
                              hipStream_t stream) {
    const float* x      = (const float*)d_in[0];
    const float* W_attn = (const float*)d_in[1];
    const float* b_attn = (const float*)d_in[2];
    const float* W_proj = (const float*)d_in[3];
    const float* b_proj = (const float*)d_in[4];
    float* out = (float*)d_out;

    // Workspace carve-up (all sizes multiples of 16B).
    char* ws = (char*)d_ws;
    __bf16* xb  = (__bf16*)ws;                 ws += (size_t)BT * CC * 2;
    __bf16* wab = (__bf16*)ws;                 ws += (size_t)C3 * CC * 2;
    __bf16* wpb = (__bf16*)ws;                 ws += (size_t)CC * CC * 2;
    __bf16* qb  = (__bf16*)ws;                 ws += (size_t)BB * HH * TT * HD * 2;
    __bf16* kb  = (__bf16*)ws;                 ws += (size_t)BB * HH * TT * HD * 2;
    __bf16* vtb = (__bf16*)ws;                 ws += (size_t)BB * HH * HD * TT * 2;
    __bf16* yb  = (__bf16*)ws;                 ws += (size_t)BT * CC * 2;

    // 1) fp32 -> bf16 conversions
    {
        int n = BT * CC;
        cvt_bf16_kernel<<<(n + 255) / 256, 256, 0, stream>>>(x, xb, n);
        n = C3 * CC;
        cvt_bf16_kernel<<<(n + 255) / 256, 256, 0, stream>>>(W_attn, wab, n);
        n = CC * CC;
        cvt_bf16_kernel<<<(n + 255) / 256, 256, 0, stream>>>(W_proj, wpb, n);
    }

    // 2) QKV projection: 256 M-tiles x 48 N-supertiles = 12288 waves
    qkv_gemm_kernel<<<(256 * 48) / 4, 128, 0, stream>>>(xb, wab, b_attn,
                                                        qb, kb, vtb);

    // 3) Flash attention: B*H*(T/16) = 4096 waves
    attn_kernel<<<(BB * HH * (TT / 16)) / 4, 128, 0, stream>>>(qb, kb, vtb, yb);

    // 4) Output projection: 256 x 16 = 4096 waves
    proj_gemm_kernel<<<(256 * 16) / 4, 128, 0, stream>>>(yb, wpb, b_proj, out);
}
